// C_GNN_18159121728197
// MI455X (gfx1250) — matrix operand order repaired and checked by reading, not compiled, not run on hardware
//
#include <hip/hip_runtime.h>
#include <hip/hip_bf16.h>
#include <math.h>

typedef __bf16 bf16;
typedef __attribute__((ext_vector_type(16))) __bf16 v16bf;
typedef __attribute__((ext_vector_type(8)))  __bf16 v8bf;
typedef __attribute__((ext_vector_type(8)))  float  v8f;

#define NNODES 16384
#define NEDGES 262144
#define NEN    (NEDGES + NNODES)
#define CIN    16
#define LIN    128
#define COUT   64
#define TPOOL  62

// ---------------------------------------------------------------------------
// WMMA fragment loaders (bf16, 16x16x32), index math per CDNA5 ISA layouts.
// A (16x32): lane l holds row M=l&15; elems 0..7 -> K = k0 + 8*(l>=16) + e,
//            elems 8..15 -> K = k0 + 16 + 8*(l>=16) + (e-8)
// B (32x16): lane l holds col N=l&15; same K mapping as A (8*(l>=16) split)
// C (16x16): vgpr v, lane l -> row = v + 8*(l>=16), col = l&15
// All strides are multiples of 8 elements so v8bf (16B) loads stay aligned.
// ---------------------------------------------------------------------------
__device__ __forceinline__ v16bf load_frag_a(const bf16* base, int row0, int stride, int k0) {
  const int lane = threadIdx.x & 31;
  const bf16* p = base + (size_t)(row0 + (lane & 15)) * stride + k0 + ((lane & 16) >> 1);
  v8bf lo = *(const v8bf*)p;
  v8bf hi = *(const v8bf*)(p + 16);
  v16bf f;
#pragma unroll
  for (int e = 0; e < 8; ++e) { f[e] = lo[e]; f[e + 8] = hi[e]; }
  return f;
}

__device__ __forceinline__ v16bf load_frag_b(const bf16* base, int col0, int stride, int k0) {
  const int lane = threadIdx.x & 31;
  const bf16* p = base + (size_t)(col0 + (lane & 15)) * stride + k0 + ((lane & 16) >> 1);
  v8bf lo = *(const v8bf*)p;
  v8bf hi = *(const v8bf*)(p + 16);
  v16bf f;
#pragma unroll
  for (int e = 0; e < 8; ++e) { f[e] = lo[e]; f[e + 8] = hi[e]; }
  return f;
}

__device__ __forceinline__ v8f wmma_bf16(v16bf a, v16bf b, v8f c) {
  return __builtin_amdgcn_wmma_f32_16x16x32_bf16(false, a, false, b, (short)0, c, false, false);
}

__device__ __forceinline__ float sigmoidf_(float x) { return 1.0f / (1.0f + __expf(-x)); }

// order-preserving float<->uint map for atomicMax-based segment max
__device__ __forceinline__ unsigned fkey(float f) {
  unsigned u = __float_as_uint(f);
  return (u & 0x80000000u) ? ~u : (u | 0x80000000u);
}
__device__ __forceinline__ float funkey(unsigned k) {
  unsigned u = (k & 0x80000000u) ? (k ^ 0x80000000u) : ~k;
  return __uint_as_float(u);
}

// ---------------------------------------------------------------------------
// Prep: pack bf16 weight matrices into workspace.
// ---------------------------------------------------------------------------
__global__ void prep_kernel(const float* __restrict__ conv_w,
                            const float* __restrict__ w_ih, const float* __restrict__ w_hh,
                            const float* __restrict__ b_ih, const float* __restrict__ b_hh,
                            const float* __restrict__ lin_w, const float* __restrict__ gat_w,
                            const float* __restrict__ lin_edge_w, const float* __restrict__ att_edge,
                            bf16* __restrict__ aconv, bf16* __restrict__ wlstm,
                            float* __restrict__ blstm, bf16* __restrict__ linw,
                            bf16* __restrict__ gatw, float* __restrict__ ce) {
  const int tid = threadIdx.x;
  // conv weights [64][96], K = ci*5+kk padded 80->96
  for (int i = tid; i < 64 * 96; i += 256) {
    int co = i / 96, k = i - co * 96;
    float v = (k < 80) ? conv_w[co * 80 + k] : 0.f;
    aconv[i] = (bf16)v;
  }
  // LSTM concat weights [256 gates][128 = x|h]
  for (int i = tid; i < 256 * 128; i += 256) {
    int g = i >> 7, c = i & 127;
    float v = (c < 64) ? w_ih[g * 64 + c] : w_hh[g * 64 + (c - 64)];
    wlstm[i] = (bf16)v;
  }
  if (tid < 256) blstm[tid] = b_ih[tid] + b_hh[tid];
  for (int i = tid; i < 64 * 64; i += 256) {
    linw[i] = (bf16)lin_w[i];
    gatw[i] = (bf16)gat_w[i];
  }
  if (tid == 0) {
    float s = 0.f;
    for (int c = 0; c < 64; ++c) s += lin_edge_w[c] * att_edge[c];
    ce[0] = s;
  }
}

__global__ void zero_kernel(unsigned* __restrict__ p, size_t n) {
  size_t i = (size_t)blockIdx.x * blockDim.x + threadIdx.x;
  if (i < n) p[i] = 0u;
}

// ---------------------------------------------------------------------------
// Conv1d(k5)+ReLU+AvgPool2 + Linear, fused. One node per 256-thread block.
// GEMM1: y[64co][128t] = A[64][96] x B[96][128] (im2col), WMMA bf16.
// Pool via lane shuffle (adjacent columns are adjacent lanes).
// GEMM2: H[62t][64o] = Xt[64][64] x lin_w^T, WMMA bf16.
// ---------------------------------------------------------------------------
__global__ __launch_bounds__(256) void conv_lin_kernel(
    const float* __restrict__ x, const float* __restrict__ conv_b,
    const float* __restrict__ lin_b, const bf16* __restrict__ aconv,
    const bf16* __restrict__ linw, float* __restrict__ cnn_out,
    bf16* __restrict__ hlin) {
  __shared__ float xs[CIN * LIN];      // 8 KB
  __shared__ bf16  Bt[128 * 104];      // 26 KB  (rows = t, cols = K, padded)
  __shared__ bf16  Aw[64 * 96];        // 12 KB
  __shared__ bf16  XtL[64 * 64];       // 8 KB   (rows = pooled t, cols = co)
  __shared__ float cb[64];
  __shared__ float lb[64];
  const int tid = threadIdx.x;
  const int n = blockIdx.x;
  const int lane = tid & 31, wave = tid >> 5;

  const float* xp = x + (size_t)n * (CIN * LIN);
  __builtin_prefetch(xp, 0, 1);
  for (int i = tid; i < CIN * LIN; i += 256) xs[i] = xp[i];
  const unsigned* aw32 = (const unsigned*)aconv;
  unsigned* Aw32 = (unsigned*)Aw;
  for (int i = tid; i < 64 * 96 / 2; i += 256) Aw32[i] = aw32[i];
  if (tid < 64) { cb[tid] = conv_b[tid]; lb[tid] = lin_b[tid]; }
  unsigned* Bt32 = (unsigned*)Bt;
  for (int i = tid; i < 128 * 104 / 2; i += 256) Bt32[i] = 0u;
  unsigned* Xt32 = (unsigned*)XtL;
  for (int i = tid; i < 64 * 64 / 2; i += 256) Xt32[i] = 0u;
  __syncthreads();

  // im2col: Bt[t][ci*5+kk] = xs[ci][t+kk], t in [0,124)
  for (int i = tid; i < 124 * 80; i += 256) {
    int t = i / 80, k = i - t * 80;
    int ci = k / 5, kk = k - ci * 5;
    Bt[t * 104 + k] = (bf16)xs[ci * LIN + t + kk];
  }
  __syncthreads();

  // GEMM1: 4 Mtiles x 8 Ntiles; wave -> (mt = w>>1, nt in 4*(w&1)..+3)
  const int mt = wave >> 1;
  const int ntb = (wave & 1) * 4;
  for (int nti = 0; nti < 4; ++nti) {
    const int nt = ntb + nti;
    v8f c = {};
#pragma unroll
    for (int kt = 0; kt < 3; ++kt) {
      v16bf a = load_frag_a(Aw, mt * 16, 96, kt * 32);
      v16bf b = load_frag_b(Bt, nt * 16, 104, kt * 32);
      c = wmma_bf16(a, b, c);
    }
    const int tcol = nt * 16 + (lane & 15);
#pragma unroll
    for (int v = 0; v < 8; ++v) {
      const int row = mt * 16 + v + ((lane & 16) >> 1);  // output channel
      float r = c[v] + cb[row];
      r = r > 0.f ? r : 0.f;                              // ReLU before pool
      float o = __shfl_xor(r, 1);
      float p = 0.5f * (r + o);
      const int j = tcol >> 1;
      if (((lane & 1) == 0) && j < TPOOL) {
        cnn_out[(size_t)n * (COUT * TPOOL) + row * TPOOL + j] = p;
        XtL[j * 64 + row] = (bf16)p;
      }
    }
  }
  __syncthreads();

  // GEMM2: H[t][o] = XtL x lin_w^T; 16 tiles, 2 per wave
  bf16* hout = hlin + (size_t)n * (TPOOL * COUT);
#pragma unroll
  for (int i = 0; i < 2; ++i) {
    const int tile = wave * 2 + i;
    const int m2 = tile >> 2, n2 = tile & 3;
    v8f c = {};
#pragma unroll
    for (int kt = 0; kt < 2; ++kt) {
      v16bf a = load_frag_a(XtL, m2 * 16, 64, kt * 32);
      v16bf b = load_frag_b(linw, n2 * 16, 64, kt * 32);
      c = wmma_bf16(a, b, c);
    }
    const int ocol = n2 * 16 + (lane & 15);
#pragma unroll
    for (int v = 0; v < 8; ++v) {
      const int trow = m2 * 16 + v + ((lane & 16) >> 1);
      if (trow < TPOOL) hout[trow * COUT + ocol] = (bf16)(c[v] + lb[ocol]);
    }
  }
}

// ---------------------------------------------------------------------------
// One LSTM step: gates[32n][256g] = [x_t | h_prev][32][128] x Wcat^T, WMMA.
// Each block owns 32 nodes (reads/writes only its own h/c -> no races).
// ---------------------------------------------------------------------------
__global__ __launch_bounds__(256) void lstm_step_kernel(
    const bf16* __restrict__ hlin, bf16* __restrict__ hcur, float* __restrict__ cst,
    const bf16* __restrict__ wlstm, const float* __restrict__ blstm, int t) {
  __shared__ bf16  As[32 * 128];   // 8 KB
  __shared__ float gsm[32 * 260];  // 33 KB
  const int tid = threadIdx.x;
  const int lane = tid & 31, wave = tid >> 5;
  const int node0 = blockIdx.x * 32;

  for (int i = tid; i < 32 * 128; i += 256) {
    const int r = i >> 7, c = i & 127;
    const int n = node0 + r;
    As[i] = (c < 64) ? hlin[(size_t)n * (TPOOL * COUT) + (size_t)t * COUT + c]
                     : hcur[(size_t)n * COUT + (c - 64)];
  }
  __syncthreads();

#pragma unroll
  for (int i = 0; i < 4; ++i) {
    const int tile = wave * 4 + i;          // 32 tiles: 2 Mtiles x 16 Ntiles
    const int m2 = tile >> 4, n2 = tile & 15;
    v8f c = {};
#pragma unroll
    for (int kt = 0; kt < 4; ++kt) {
      v16bf a = load_frag_a(As, m2 * 16, 128, kt * 32);
      v16bf b = load_frag_b(wlstm, n2 * 16, 128, kt * 32);  // straight from L2
      c = wmma_bf16(a, b, c);
    }
    const int g = n2 * 16 + (lane & 15);
#pragma unroll
    for (int v = 0; v < 8; ++v) {
      const int r = m2 * 16 + v + ((lane & 16) >> 1);
      gsm[r * 260 + g] = c[v];
    }
  }
  __syncthreads();

  for (int u = tid; u < 32 * 64; u += 256) {
    const int r = u >> 6, cc = u & 63;
    const int n = node0 + r;
    const float iv = gsm[r * 260 + cc]       + blstm[cc];
    const float fv = gsm[r * 260 + 64 + cc]  + blstm[64 + cc];
    const float gv = gsm[r * 260 + 128 + cc] + blstm[128 + cc];
    const float ov = gsm[r * 260 + 192 + cc] + blstm[192 + cc];
    const float cold = cst[(size_t)n * 64 + cc];
    const float cn = sigmoidf_(fv) * cold + sigmoidf_(iv) * tanhf(gv);
    const float hn = sigmoidf_(ov) * tanhf(cn);
    cst[(size_t)n * 64 + cc] = cn;
    hcur[(size_t)n * 64 + cc] = (bf16)hn;
  }
}

// ---------------------------------------------------------------------------
// GAT node transform: xl = h_final x gat_w^T (WMMA) + attention dots.
// ---------------------------------------------------------------------------
__global__ __launch_bounds__(256) void gat_transform_kernel(
    const bf16* __restrict__ hcur, const bf16* __restrict__ gatw,
    const float* __restrict__ att_src, const float* __restrict__ att_dst,
    float* __restrict__ xl, float* __restrict__ a_src, float* __restrict__ a_dst) {
  __shared__ bf16  As[64 * 64];    // 8 KB
  __shared__ float xsm[64 * 66];   // 17 KB
  const int tid = threadIdx.x;
  const int lane = tid & 31, wave = tid >> 5;
  const int node0 = blockIdx.x * 64;

  const unsigned* hp = (const unsigned*)(hcur + (size_t)node0 * 64);
  unsigned* As32 = (unsigned*)As;
  for (int i = tid; i < 64 * 64 / 2; i += 256) As32[i] = hp[i];
  __syncthreads();

#pragma unroll
  for (int i = 0; i < 2; ++i) {
    const int tile = wave * 2 + i;          // 16 tiles: 4x4
    const int m2 = tile >> 2, n2 = tile & 3;
    v8f c = {};
#pragma unroll
    for (int kt = 0; kt < 2; ++kt) {
      v16bf a = load_frag_a(As, m2 * 16, 64, kt * 32);
      v16bf b = load_frag_b(gatw, n2 * 16, 64, kt * 32);
      c = wmma_bf16(a, b, c);
    }
    const int o = n2 * 16 + (lane & 15);
#pragma unroll
    for (int v = 0; v < 8; ++v) {
      const int r = m2 * 16 + v + ((lane & 16) >> 1);
      xsm[r * 66 + o] = c[v];
    }
  }
  __syncthreads();

  for (int i = tid; i < 64 * 64; i += 256)
    xl[(size_t)node0 * 64 + i] = xsm[(i >> 6) * 66 + (i & 63)];
  if (tid < 64) {
    float ss = 0.f, sd = 0.f;
    for (int o = 0; o < 64; ++o) {
      const float v = xsm[tid * 66 + o];
      ss += v * att_src[o];
      sd += v * att_dst[o];
    }
    a_src[node0 + tid] = ss;
    a_dst[node0 + tid] = sd;
  }
}

// ---------------------------------------------------------------------------
// GAT edge-side kernels.
// ---------------------------------------------------------------------------
__global__ void deg_kernel(const int* __restrict__ dst, const float* __restrict__ w,
                           float* __restrict__ deg, float* __restrict__ wsum) {
  const int e = blockIdx.x * blockDim.x + threadIdx.x;
  if (e < NEDGES) {
    const int d = dst[e];
    atomicAdd(&deg[d], 1.0f);
    atomicAdd(&wsum[d], w[e]);
  }
}

__global__ void loopattr_kernel(const float* __restrict__ deg, const float* __restrict__ wsum,
                                float* __restrict__ lattr) {
  const int n = blockIdx.x * blockDim.x + threadIdx.x;
  if (n < NNODES) lattr[n] = wsum[n] / fmaxf(deg[n], 1.0f);
}

__global__ void alpha_kernel(const int* __restrict__ src, const int* __restrict__ dst,
                             const float* __restrict__ ew, const float* __restrict__ lattr,
                             const float* __restrict__ a_src, const float* __restrict__ a_dst,
                             const float* __restrict__ ce, float* __restrict__ alpha,
                             unsigned* __restrict__ amax) {
  const int e = blockIdx.x * blockDim.x + threadIdx.x;
  if (e >= NEN) return;
  int s, d; float ea;
  if (e < NEDGES) { s = src[e]; d = dst[e]; ea = ew[e]; }
  else            { s = d = e - NEDGES;     ea = lattr[s]; }
  float a = a_src[s] + a_dst[d] + ea * ce[0];
  a = (a > 0.f) ? a : 0.2f * a;   // leaky relu
  alpha[e] = a;
  atomicMax(&amax[d], fkey(a));
}

__global__ void softmax_kernel(const int* __restrict__ dst, float* __restrict__ alpha,
                               const unsigned* __restrict__ amax, float* __restrict__ denom) {
  const int e = blockIdx.x * blockDim.x + threadIdx.x;
  if (e >= NEN) return;
  const int d = (e < NEDGES) ? dst[e] : (e - NEDGES);
  const float ex = __expf(alpha[e] - funkey(amax[d]));
  alpha[e] = ex;
  atomicAdd(&denom[d], ex);
}

__global__ void scatter_kernel(const int* __restrict__ src, const int* __restrict__ dst,
                               const float* __restrict__ alpha, const float* __restrict__ denom,
                               const float* __restrict__ xl, float* __restrict__ outreg,
                               float* __restrict__ score_out) {
  const int e = blockIdx.x * 4 + (threadIdx.x >> 6);
  const int c = threadIdx.x & 63;
  if (e >= NEN) return;
  int s, d;
  if (e < NEDGES) { s = src[e]; d = dst[e]; } else { s = d = e - NEDGES; }
  const float sc = alpha[e] / denom[d];
  atomicAdd(&outreg[(size_t)d * 64 + c], sc * xl[(size_t)s * 64 + c]);
  if (c == 0) score_out[e] = sc;
}

__global__ void finalize_kernel(float* __restrict__ outreg, const float* __restrict__ gb) {
  const int i = blockIdx.x * blockDim.x + threadIdx.x;
  if (i < NNODES * 64) outreg[i] += gb[i & 63];
}

// ---------------------------------------------------------------------------
// Workspace layout
// ---------------------------------------------------------------------------
static constexpr size_t alup(size_t x) { return (x + 255) & ~(size_t)255; }
static constexpr size_t OFF_ACONV = 0;
static constexpr size_t OFF_WLSTM = alup(OFF_ACONV + 64 * 96 * 2);
static constexpr size_t OFF_BLSTM = alup(OFF_WLSTM + 256 * 128 * 2);
static constexpr size_t OFF_CE    = alup(OFF_BLSTM + 256 * 4);
static constexpr size_t OFF_LINW  = alup(OFF_CE + 256);
static constexpr size_t OFF_GATW  = alup(OFF_LINW + 64 * 64 * 2);
static constexpr size_t OFF_HLIN  = alup(OFF_GATW + 64 * 64 * 2);
static constexpr size_t OFF_HCUR  = alup(OFF_HLIN + (size_t)NNODES * TPOOL * COUT * 2);
static constexpr size_t OFF_CST   = alup(OFF_HCUR + (size_t)NNODES * 64 * 2);
static constexpr size_t OFF_XL    = alup(OFF_CST + (size_t)NNODES * 64 * 4);
static constexpr size_t OFF_ASRC  = alup(OFF_XL + (size_t)NNODES * 64 * 4);
static constexpr size_t OFF_ADST  = alup(OFF_ASRC + (size_t)NNODES * 4);
static constexpr size_t OFF_DEG   = alup(OFF_ADST + (size_t)NNODES * 4);
static constexpr size_t OFF_WSUM  = alup(OFF_DEG + (size_t)NNODES * 4);
static constexpr size_t OFF_LATTR = alup(OFF_WSUM + (size_t)NNODES * 4);
static constexpr size_t OFF_AMAX  = alup(OFF_LATTR + (size_t)NNODES * 4);
static constexpr size_t OFF_DENOM = alup(OFF_AMAX + (size_t)NNODES * 4);
static constexpr size_t OFF_ALPHA = alup(OFF_DENOM + (size_t)NNODES * 4);
static constexpr size_t WS_NEED   = alup(OFF_ALPHA + (size_t)NEN * 4);

extern "C" void kernel_launch(void* const* d_in, const int* in_sizes, int n_in,
                              void* d_out, int out_size, void* d_ws, size_t ws_size,
                              hipStream_t stream) {
  (void)in_sizes; (void)n_in; (void)out_size;
  if (ws_size < WS_NEED) return;

  const float* x          = (const float*)d_in[0];
  const int*   ei         = (const int*)d_in[1];
  const float* ew         = (const float*)d_in[2];
  const float* conv_w     = (const float*)d_in[3];
  const float* conv_b     = (const float*)d_in[4];
  const float* lin_w      = (const float*)d_in[5];
  const float* lin_b      = (const float*)d_in[6];
  const float* w_ih       = (const float*)d_in[7];
  const float* w_hh       = (const float*)d_in[8];
  const float* b_ih       = (const float*)d_in[9];
  const float* b_hh       = (const float*)d_in[10];
  const float* gat_w      = (const float*)d_in[11];
  const float* att_src    = (const float*)d_in[12];
  const float* att_dst    = (const float*)d_in[13];
  const float* lin_edge_w = (const float*)d_in[14];
  const float* att_edge   = (const float*)d_in[15];
  const float* gat_bias   = (const float*)d_in[16];
  const int* e_src = ei;
  const int* e_dst = ei + NEDGES;

  char* w = (char*)d_ws;
  bf16*  aconv = (bf16*)(w + OFF_ACONV);
  bf16*  wlstm = (bf16*)(w + OFF_WLSTM);
  float* blstm = (float*)(w + OFF_BLSTM);
  float* ce    = (float*)(w + OFF_CE);
  bf16*  linw  = (bf16*)(w + OFF_LINW);
  bf16*  gatw  = (bf16*)(w + OFF_GATW);
  bf16*  hlin  = (bf16*)(w + OFF_HLIN);
  bf16*  hcur  = (bf16*)(w + OFF_HCUR);
  float* cst   = (float*)(w + OFF_CST);
  float* xl    = (float*)(w + OFF_XL);
  float* asrc  = (float*)(w + OFF_ASRC);
  float* adst  = (float*)(w + OFF_ADST);
  float* deg   = (float*)(w + OFF_DEG);
  float* wsum  = (float*)(w + OFF_WSUM);
  float* lattr = (float*)(w + OFF_LATTR);
  unsigned* amax = (unsigned*)(w + OFF_AMAX);
  float* denom = (float*)(w + OFF_DENOM);
  float* alpha = (float*)(w + OFF_ALPHA);

  float* out_cnn   = (float*)d_out;                                   // [N,64,62]
  float* out_gat   = out_cnn + (size_t)NNODES * COUT * TPOOL;         // [N,64]
  float* out_score = out_gat + (size_t)NNODES * COUT;                 // [E+N]

  prep_kernel<<<1, 256, 0, stream>>>(conv_w, w_ih, w_hh, b_ih, b_hh, lin_w, gat_w,
                                     lin_edge_w, att_edge, aconv, wlstm, blstm,
                                     linw, gatw, ce);

  auto zero = [&](void* p, size_t n_u32) {
    zero_kernel<<<(unsigned)((n_u32 + 255) / 256), 256, 0, stream>>>((unsigned*)p, n_u32);
  };
  zero(hcur,  (size_t)NNODES * 64 * 2 / 4);
  zero(cst,   (size_t)NNODES * 64);
  zero(deg,   NNODES);
  zero(wsum,  NNODES);
  zero(amax,  NNODES);
  zero(denom, NNODES);
  zero(out_gat, (size_t)NNODES * 64);

  conv_lin_kernel<<<NNODES, 256, 0, stream>>>(x, conv_b, lin_b, aconv, linw,
                                              out_cnn, hlin);

  for (int t = 0; t < TPOOL; ++t)
    lstm_step_kernel<<<NNODES / 32, 256, 0, stream>>>(hlin, hcur, cst, wlstm, blstm, t);

  gat_transform_kernel<<<NNODES / 64, 256, 0, stream>>>(hcur, gatw, att_src, att_dst,
                                                        xl, asrc, adst);

  deg_kernel<<<NEDGES / 256, 256, 0, stream>>>(e_dst, ew, deg, wsum);
  loopattr_kernel<<<NNODES / 256, 256, 0, stream>>>(deg, wsum, lattr);
  alpha_kernel<<<(NEN + 255) / 256, 256, 0, stream>>>(e_src, e_dst, ew, lattr, asrc, adst,
                                                      ce, alpha, amax);
  softmax_kernel<<<(NEN + 255) / 256, 256, 0, stream>>>(e_dst, alpha, amax, denom);
  scatter_kernel<<<(NEN + 3) / 4, 256, 0, stream>>>(e_src, e_dst, alpha, denom, xl,
                                                    out_gat, out_score);
  finalize_kernel<<<(NNODES * 64) / 256, 256, 0, stream>>>(out_gat, gat_bias);
}
